// RNNQFunction_28415503630465
// MI455X (gfx1250) — matrix operand order repaired and checked by reading, not compiled, not run on hardware
//
#include <hip/hip_runtime.h>

typedef __attribute__((ext_vector_type(2)))  float    v2f;
typedef __attribute__((ext_vector_type(8)))  float    v8f;
typedef __attribute__((ext_vector_type(16))) _Float16 v16h;
typedef __attribute__((ext_vector_type(2)))  __fp16   v2fp16;

namespace {
constexpr int kB = 4096, kT = 512, kC = 4, kH = 32, kO = 3;
constexpr int kChains = 2;                       // independent batch tiles per wave (ILP)
constexpr int kTiles  = kB / 16;                 // 256
constexpr int kWaves  = kTiles / kChains;        // 128
constexpr int kBlock  = 128;                     // 4 waves per block
constexpr int kGrid   = (kWaves * 32) / kBlock;  // 32

__device__ inline unsigned pk16(float a, float b) {
  v2fp16 p = __builtin_amdgcn_cvt_pkrtz(a, b);   // v_cvt_pk_rtz_f16_f32
  return __builtin_bit_cast(unsigned, p);
}
// packed-f16 ReLU: one VOP3P op for two elements (avoids f32 canonicalize+max pair)
__device__ inline unsigned relu_pk(unsigned x) {
  unsigned r;
  asm("v_pk_max_num_f16 %0, %1, 0" : "=v"(r) : "v"(x));
  return r;
}
__device__ inline unsigned swap16(unsigned x) {
  // ds_swizzle_b32 group-of-32: xor_mask=0x10, and_mask=0x1f -> half-wave swap
  return (unsigned)__builtin_amdgcn_ds_swizzle((int)x, 0x401F);
}
__device__ inline float swap16f(float x) {
  return __builtin_bit_cast(float, swap16(__builtin_bit_cast(unsigned, x)));
}
} // namespace

union HPack { unsigned u[8]; v16h v; };

__global__ __launch_bounds__(kBlock) void rnn_relu_scan(
    const float* __restrict__ seq,  const float* __restrict__ w_ih,
    const float* __restrict__ w_hh, const float* __restrict__ b_ih,
    const float* __restrict__ b_hh, const float* __restrict__ fc_w,
    const float* __restrict__ fc_b, float* __restrict__ out)
{
  const int lane = (int)(threadIdx.x & 31);
  const int wave = (int)((blockIdx.x * blockDim.x + threadIdx.x) >> 5);
  const int m    = lane & 15;     // row/col index within 16
  const int hi   = lane >> 4;     // which half-wave
  const bool lo  = (hi == 0);

  // ---- W_hh as two constant f16 A-tiles (16x32, M=n rows 0..15 / 16..31) ----
  // A layout: lane(m + 16*hk) VGPR v holds K = 2v + 16*hk, packed pairs.
  v16h a_hh0, a_hh1;
  {
    HPack p0, p1;
#pragma unroll
    for (int v = 0; v < 8; ++v) {
      const int k = 2 * v + 16 * hi;
      p0.u[v] = pk16(w_hh[m * kH + k],          w_hh[m * kH + k + 1]);
      p1.u[v] = pk16(w_hh[(m + 16) * kH + k],   w_hh[(m + 16) * kH + k + 1]);
    }
    a_hh0 = p0.v; a_hh1 = p1.v;
  }

  // ---- W_ih as two constant f32 A-tiles (16x4): VGPR0=K(2hi), VGPR1=K(2hi+1) ----
  v2f a_ih0, a_ih1;
  a_ih0.x = w_ih[m * kC + 2 * hi];          a_ih0.y = w_ih[m * kC + 2 * hi + 1];
  a_ih1.x = w_ih[(m + 16) * kC + 2 * hi];   a_ih1.y = w_ih[(m + 16) * kC + 2 * hi + 1];

  // ---- bias broadcast as C-tiles (element (M=n, N=b) = b_ih[n]+b_hh[n]) ----
  v8f cb0, cb1;
#pragma unroll
  for (int r = 0; r < 8; ++r) {
    const int n = r + 8 * hi;
    cb0[r] = b_ih[n]      + b_hh[n];
    cb1[r] = b_ih[n + 16] + b_hh[n + 16];
  }

  // ---- per-chain state: seq pointer + h^T (f16 B-operand), h0 = 0 ----
  const float* sp[kChains];
  HPack bh[kChains];
#pragma unroll
  for (int c = 0; c < kChains; ++c) {
    const int b = (wave * kChains + c) * 16 + m;   // batch row handled by this lane pair
    sp[c] = seq + (size_t)b * kT * kC + 2 * hi;    // lanes 0-15: channels 0,1; 16-31: 2,3
#pragma unroll
    for (int v = 0; v < 8; ++v) bh[c].u[v] = 0u;
  }

  v2f bs[kChains];
#pragma unroll
  for (int c = 0; c < kChains; ++c) { bs[c].x = sp[c][0]; bs[c].y = sp[c][1]; } // t=0

  v8f d0[kChains], d1[kChains];

  for (int t = 0; t < kT; ++t) {
    // prefetch next timestep's seq slice (B operand of the K=4 f32 WMMA)
    const int tn = (t + 1 < kT) ? (t + 1) : (kT - 1);
    v2f bn[kChains];
#pragma unroll
    for (int c = 0; c < kChains; ++c) {
      const float* p = sp[c] + (size_t)tn * kC;
      bn[c].x = p[0]; bn[c].y = p[1];
    }

    // input projection: D = W_ih * x_t^T + bias   (full fp32, K=4)
#pragma unroll
    for (int c = 0; c < kChains; ++c) {
      d0[c] = __builtin_amdgcn_wmma_f32_16x16x4_f32(false, a_ih0, false, bs[c],
                                                    (short)0, cb0, false, false);
      d1[c] = __builtin_amdgcn_wmma_f32_16x16x4_f32(false, a_ih1, false, bs[c],
                                                    (short)0, cb1, false, false);
    }
    // recurrence: D += W_hh * h_{t-1}^T           (f16 in, fp32 accumulate, K=32)
#pragma unroll
    for (int c = 0; c < kChains; ++c) {
      d0[c] = __builtin_amdgcn_wmma_f32_16x16x32_f16(false, a_hh0, false, bh[c].v,
                                                     (short)0, d0[c], false, false);
      d1[c] = __builtin_amdgcn_wmma_f32_16x16x32_f16(false, a_hh1, false, bh[c].v,
                                                     (short)0, d1[c], false, false);
    }
    // f32->f16 pack, packed ReLU, then D-layout -> B-layout (half-wave swap)
#pragma unroll
    for (int c = 0; c < kChains; ++c) {
      unsigned P0[4], P1[4];
#pragma unroll
      for (int j = 0; j < 4; ++j) {
        P0[j] = relu_pk(pk16(d0[c][2 * j], d0[c][2 * j + 1]));
        P1[j] = relu_pk(pk16(d1[c][2 * j], d1[c][2 * j + 1]));
      }
#pragma unroll
      for (int j = 0; j < 4; ++j) {
        const unsigned q = lo ? P1[j] : P0[j];
        const unsigned s = swap16(q);
        bh[c].u[j]     = lo ? P0[j] : s;   // K = 2j,2j+1   (+16 on hi half)
        bh[c].u[4 + j] = lo ? s : P1[j];   // K = 8+2j,...  (+16 on hi half)
      }
      bs[c] = bn[c];
    }
  }

  // ---- head: out[b][o] = fc_b[o] + sum_n fc_w[o][n] * relu(h_last[n][b]) ----
#pragma unroll
  for (int c = 0; c < kChains; ++c) {
    float acc0 = 0.f, acc1 = 0.f, acc2 = 0.f;
#pragma unroll
    for (int r = 0; r < 8; ++r) {
      const float h0 = fmaxf(d0[c][r], 0.f);
      const float h1 = fmaxf(d1[c][r], 0.f);
      const int n0 = r + 8 * hi, n1 = n0 + 16;
      acc0 += fc_w[0 * kH + n0] * h0 + fc_w[0 * kH + n1] * h1;
      acc1 += fc_w[1 * kH + n0] * h0 + fc_w[1 * kH + n1] * h1;
      acc2 += fc_w[2 * kH + n0] * h0 + fc_w[2 * kH + n1] * h1;
    }
    // lane l and lane l^16 hold the same batch column: reduce across halves
    acc0 += swap16f(acc0);
    acc1 += swap16f(acc1);
    acc2 += swap16f(acc2);
    if (lo) {
      const int b = (wave * kChains + c) * 16 + m;
      out[b * kO + 0] = acc0 + fc_b[0];
      out[b * kO + 1] = acc1 + fc_b[1];
      out[b * kO + 2] = acc2 + fc_b[2];
    }
  }
}

extern "C" void kernel_launch(void* const* d_in, const int* in_sizes, int n_in,
                              void* d_out, int out_size, void* d_ws, size_t ws_size,
                              hipStream_t stream) {
  const float* seq  = (const float*)d_in[0];
  const float* w_ih = (const float*)d_in[1];
  const float* w_hh = (const float*)d_in[2];
  const float* b_ih = (const float*)d_in[3];
  const float* b_hh = (const float*)d_in[4];
  const float* fc_w = (const float*)d_in[5];
  const float* fc_b = (const float*)d_in[6];
  float* out = (float*)d_out;
  (void)in_sizes; (void)n_in; (void)out_size; (void)d_ws; (void)ws_size;
  rnn_relu_scan<<<kGrid, kBlock, 0, stream>>>(seq, w_ih, w_hh, b_ih, b_hh, fc_w, fc_b, out);
}
